// NeuronLongCatTransformer_54743653155410
// MI455X (gfx1250) — matrix operand order repaired and checked by reading, not compiled, not run on hardware
//
#include <hip/hip_runtime.h>
#include <hip/hip_bf16.h>

typedef __attribute__((ext_vector_type(8)))  float  v8f;
typedef __attribute__((ext_vector_type(4)))  float  v4f;
typedef __attribute__((ext_vector_type(16))) __bf16 v16bf;
typedef __attribute__((ext_vector_type(8)))  __bf16 v8bf;

#define HEADS   24
#define HD      128
#define DIM     3072
#define S_IMG   2048
#define S_TXT   512
#define SEQ     2560
#define TM      128
#define TK      32
#define BR      64
#define BC      32
#define ATT_SCALE 0.08838834764831845f
#define EPS 1e-6f

__device__ __forceinline__ v8f vzero8() {
    v8f z;
#pragma unroll
    for (int e = 0; e < 8; e++) z[e] = 0.0f;
    return z;
}

__device__ __forceinline__ v16bf combine16(v8bf lo, v8bf hi) {
    v16bf f;
#pragma unroll
    for (int e = 0; e < 8; e++) { f[e] = lo[e]; f[8 + e] = hi[e]; }
    return f;
}

// C[M,3072] = X[M,3072] @ W[3072,3072]^T + bias, tiled 128x128, one head per N-tile.
// mode 1: fused RMS-norm + RoPE -> bf16 head-major [h][seq][d]   (Q, K)
// mode 2: bias only              -> bf16 head-major              (V)
// mode 3: bias only              -> fp32 row-major [m][3072]     (output proj)
__global__ void __launch_bounds__(256)
gemm_proj(const float* __restrict__ Xf, const __bf16* __restrict__ Xb,
          const float* __restrict__ W, const float* __restrict__ bias,
          int mode,
          __bf16* __restrict__ outB, float* __restrict__ outF,
          const float* __restrict__ cosT, const float* __restrict__ sinT,
          const float* __restrict__ normW, int seqOffset)
{
    __shared__ __bf16 Xs[TM][TK];       // 8 KB
    __shared__ __bf16 Ws[TM][TK];       // 8 KB
    __shared__ __bf16 Cbf[TM][TM];      // 32 KB (mode 1 epilogue)
    __shared__ float  rms[TM];

    const int m0 = blockIdx.x * TM;
    const int h  = blockIdx.y;          // head == N-tile
    const int n0 = h * TM;
    const int t = threadIdx.x;
    const int wave = t >> 5, lane = t & 31;
    const int wm = (wave & 3) * 32;     // wave row base
    const int wn = (wave >> 2) * 64;    // wave col base
    const int alr = lane & 15, ahi = lane >> 4;

    v8f acc[2][4];
#pragma unroll
    for (int i = 0; i < 2; i++)
#pragma unroll
        for (int j = 0; j < 4; j++) acc[i][j] = vzero8();

    for (int kt = 0; kt < DIM; kt += TK) {
        __syncthreads();
        // stage X (fp32->bf16 or straight bf16) and W (fp32->bf16) into LDS
#pragma unroll
        for (int i = 0; i < 2; i++) {
            int vec = t * 2 + i;                 // 512 vectors of 8
            int r  = vec >> 2;
            int kc = (vec & 3) << 3;
            if (Xb) {
                *(v8bf*)&Xs[r][kc] = *(const v8bf*)(Xb + (size_t)(m0 + r) * DIM + kt + kc);
            } else {
                const float* src = Xf + (size_t)(m0 + r) * DIM + kt + kc;
                v4f a = *(const v4f*)src;
                v4f b = *(const v4f*)(src + 4);
                v8bf xv;
#pragma unroll
                for (int e = 0; e < 4; e++) { xv[e] = (__bf16)a[e]; xv[4 + e] = (__bf16)b[e]; }
                *(v8bf*)&Xs[r][kc] = xv;
            }
            const float* wsrc = W + (size_t)(n0 + r) * DIM + kt + kc;
            v4f wa = *(const v4f*)wsrc;
            v4f wb = *(const v4f*)(wsrc + 4);
            v8bf wv;
#pragma unroll
            for (int e = 0; e < 4; e++) { wv[e] = (__bf16)wa[e]; wv[4 + e] = (__bf16)wb[e]; }
            *(v8bf*)&Ws[r][kc] = wv;
        }
        __syncthreads();

        // A fragments: 16x32, lane row = alr, contiguous K run k0..k0+15
        v16bf afrag[2];
#pragma unroll
        for (int i = 0; i < 2; i++) {
            int row = wm + i * 16 + alr;
            int k0  = ahi ? 16 : 0;
            afrag[i] = *(const v16bf*)&Xs[row][k0];
        }
        // B fragments: B[k][n] = W[n][k]; lane column = contiguous K run in Ws row
        v16bf bfrag[4];
#pragma unroll
        for (int j = 0; j < 4; j++) {
            int col = wn + j * 16 + alr;
            bfrag[j] = *(const v16bf*)&Ws[col][ahi ? 16 : 0];
        }
#pragma unroll
        for (int i = 0; i < 2; i++)
#pragma unroll
            for (int j = 0; j < 4; j++)
                acc[i][j] = __builtin_amdgcn_wmma_f32_16x16x32_bf16(
                    false, afrag[i], false, bfrag[j], (short)0, acc[i][j], false, false);
    }

    // ---- epilogue ----
    const int chi = ahi * 8;
    if (mode == 3) {
#pragma unroll
        for (int i = 0; i < 2; i++)
#pragma unroll
            for (int j = 0; j < 4; j++) {
                int col = wn + j * 16 + alr;
                float bv = bias[n0 + col];
#pragma unroll
                for (int e = 0; e < 8; e++) {
                    int row = wm + i * 16 + chi + e;
                    outF[(size_t)(m0 + row) * DIM + n0 + col] = acc[i][j][e] + bv;
                }
            }
        return;
    }
    if (mode == 2) {
#pragma unroll
        for (int i = 0; i < 2; i++)
#pragma unroll
            for (int j = 0; j < 4; j++) {
                int col = wn + j * 16 + alr;
                float bv = bias[n0 + col];
#pragma unroll
                for (int e = 0; e < 8; e++) {
                    int row = wm + i * 16 + chi + e;
                    size_t s = (size_t)seqOffset + m0 + row;
                    outB[((size_t)h * SEQ + s) * HD + col] = (__bf16)(acc[i][j][e] + bv);
                }
            }
        return;
    }
    // mode 1: RMS-norm over head_dim (== this 128-col tile) + RoPE
#pragma unroll
    for (int i = 0; i < 2; i++)
#pragma unroll
        for (int j = 0; j < 4; j++) {
            int col = wn + j * 16 + alr;
            float bv = bias[n0 + col];
#pragma unroll
            for (int e = 0; e < 8; e++)
                Cbf[wm + i * 16 + chi + e][col] = (__bf16)(acc[i][j][e] + bv);
        }
    __syncthreads();
    if (t < TM) {
        float ssq = 0.f;
        for (int c = 0; c < HD; c++) { float v = (float)Cbf[t][c]; ssq += v * v; }
        rms[t] = rsqrtf(ssq * (1.0f / HD) + EPS);
    }
    __syncthreads();
    {
        int row = t >> 1;
        int cb  = (t & 1) * 64;
        int pos = m0 + row;                       // position within its stream
        float r = rms[row];
        size_t orow = ((size_t)h * SEQ + (size_t)seqOffset + m0 + row) * HD;
        const float* cp = cosT + (size_t)pos * HD;
        const float* sp = sinT + (size_t)pos * HD;
#pragma unroll
        for (int i = 0; i < 32; i++) {
            int d0 = cb + 2 * i, d1 = d0 + 1;
            float x0 = (float)Cbf[row][d0] * r * normW[d0];
            float x1 = (float)Cbf[row][d1] * r * normW[d1];
            outB[orow + d0] = (__bf16)(x0 * cp[d0] - x1 * sp[d0]);
            outB[orow + d1] = (__bf16)(x1 * cp[d1] + x0 * sp[d1]);
        }
    }
}

// Flash attention: per (64-query block, head). Online softmax, WMMA for QK^T and PV.
__global__ void __launch_bounds__(256)
attn_fa(const __bf16* __restrict__ Q, const __bf16* __restrict__ K,
        const __bf16* __restrict__ V, __bf16* __restrict__ attnOut)
{
    __shared__ __bf16 Qs[BR][HD];       // 16 KB
    __shared__ __bf16 Ks[BC][HD];       // 8 KB
    __shared__ __bf16 Vt[HD][BC];       // 8 KB (transposed V)
    __shared__ float  Ss[BR][BC];       // 8 KB
    __shared__ __bf16 Ps[BR][BC];       // 4 KB
    __shared__ float  mS[BR], lS[BR], aS[BR];
    __shared__ float  part[BR][4];

    const int qi = blockIdx.x, h = blockIdx.y;
    const int t = threadIdx.x, wave = t >> 5, lane = t & 31;
    const int alr = lane & 15, ahi = lane >> 4;
    const int wm = (wave & 3) * 16;     // S/O row base
    const int sc = (wave >> 2) * 16;    // S col base
    const int on = (wave >> 2) * 64;    // O col base
    const size_t hbase = (size_t)h * SEQ * HD;

#pragma unroll
    for (int i = 0; i < 4; i++) {       // load Q tile: 64x128 halves
        int vec = t * 4 + i;
        int r = vec >> 4, d0 = (vec & 15) << 3;
        *(v8bf*)&Qs[r][d0] = *(const v8bf*)(Q + hbase + (size_t)(qi * BR + r) * HD + d0);
    }
    if (t < BR) { mS[t] = -1e30f; lS[t] = 0.f; }
    v8f o[4];
#pragma unroll
    for (int j = 0; j < 4; j++) o[j] = vzero8();
    __syncthreads();

    for (int kb = 0; kb < SEQ / BC; kb++) {
#pragma unroll
        for (int i = 0; i < 2; i++) {   // stage K (row-major) and V (transposed)
            int vec = t * 2 + i;
            int r = vec >> 4, d0 = (vec & 15) << 3;
            *(v8bf*)&Ks[r][d0] = *(const v8bf*)(K + hbase + (size_t)(kb * BC + r) * HD + d0);
            v8bf vv = *(const v8bf*)(V + hbase + (size_t)(kb * BC + r) * HD + d0);
#pragma unroll
            for (int e = 0; e < 8; e++) Vt[d0 + e][r] = vv[e];
        }
        __syncthreads();

        // S = scale * Q K^T  (one 16x16 fragment per wave)
        v8f s = vzero8();
#pragma unroll
        for (int ks = 0; ks < 4; ks++) {
            int row = wm + alr;
            int k0  = ks * 32 + (ahi ? 16 : 0);
            v16bf af = *(const v16bf*)&Qs[row][k0];
            int col = sc + alr;
            v16bf bf_ = *(const v16bf*)&Ks[col][ks * 32 + (ahi ? 16 : 0)];
            s = __builtin_amdgcn_wmma_f32_16x16x32_bf16(false, af, false, bf_, (short)0, s, false, false);
        }
#pragma unroll
        for (int e = 0; e < 8; e++)
            Ss[wm + ahi * 8 + e][sc + alr] = s[e] * ATT_SCALE;
        __syncthreads();

        // online softmax over this 32-key chunk
        {
            int r = t >> 2, p = t & 3;
            float mx = -1e30f;
            for (int c = p * 8; c < p * 8 + 8; c++) mx = fmaxf(mx, Ss[r][c]);
            part[r][p] = mx;
        }
        __syncthreads();
        if (t < BR) {
            float mx = fmaxf(fmaxf(part[t][0], part[t][1]), fmaxf(part[t][2], part[t][3]));
            float newm = fmaxf(mS[t], mx);
            aS[t] = __expf(mS[t] - newm);
            mS[t] = newm;
        }
        __syncthreads();
        {
            int r = t >> 2, p = t & 3;
            float m = mS[r], sum = 0.f;
            for (int c = p * 8; c < p * 8 + 8; c++) {
                float pv = __expf(Ss[r][c] - m);
                Ps[r][c] = (__bf16)pv;
                sum += pv;
            }
            part[r][p] = sum;
        }
        __syncthreads();
        if (t < BR)
            lS[t] = lS[t] * aS[t] + part[t][0] + part[t][1] + part[t][2] + part[t][3];

        // rescale O and accumulate P @ V
        float af8[8];
#pragma unroll
        for (int e = 0; e < 8; e++) af8[e] = aS[wm + ahi * 8 + e];
#pragma unroll
        for (int j = 0; j < 4; j++)
#pragma unroll
            for (int e = 0; e < 8; e++) o[j][e] *= af8[e];

        {
            int row = wm + alr;
            int k0  = ahi ? 16 : 0;
            v16bf pa = *(const v16bf*)&Ps[row][k0];
#pragma unroll
            for (int j = 0; j < 4; j++) {
                int col = on + j * 16 + alr;
                v16bf vb = *(const v16bf*)&Vt[col][ahi ? 16 : 0];
                o[j] = __builtin_amdgcn_wmma_f32_16x16x32_bf16(false, pa, false, vb, (short)0, o[j], false, false);
            }
        }
        __syncthreads();   // before next chunk overwrites Ks/Vt/Ps/aS
    }

    if (t < BR) aS[t] = 1.0f / lS[t];
    __syncthreads();
    float inv8[8];
#pragma unroll
    for (int e = 0; e < 8; e++) inv8[e] = aS[wm + ahi * 8 + e];
#pragma unroll
    for (int j = 0; j < 4; j++)
#pragma unroll
        for (int e = 0; e < 8; e++) {
            int row = qi * BR + wm + ahi * 8 + e;
            int col = on + j * 16 + alr;
            attnOut[(size_t)row * DIM + h * HD + col] = (__bf16)(o[j][e] * inv8[e]);
        }
}

extern "C" void kernel_launch(void* const* d_in, const int* in_sizes, int n_in,
                              void* d_out, int out_size, void* d_ws, size_t ws_size,
                              hipStream_t stream) {
    const float* hidden  = (const float*)d_in[0];
    const float* enc     = (const float*)d_in[1];
    const float* img_cos = (const float*)d_in[2];
    const float* img_sin = (const float*)d_in[3];
    const float* txt_cos = (const float*)d_in[4];
    const float* txt_sin = (const float*)d_in[5];
    const float* Wq  = (const float*)d_in[6];
    const float* Wk  = (const float*)d_in[7];
    const float* Wv  = (const float*)d_in[8];
    const float* Waq = (const float*)d_in[9];
    const float* Wak = (const float*)d_in[10];
    const float* Wav = (const float*)d_in[11];
    const float* Wo  = (const float*)d_in[12];
    const float* Wao = (const float*)d_in[13];
    const float* bq  = (const float*)d_in[14];
    const float* bk  = (const float*)d_in[15];
    const float* bv  = (const float*)d_in[16];
    const float* baq = (const float*)d_in[17];
    const float* bak = (const float*)d_in[18];
    const float* bav = (const float*)d_in[19];
    const float* bo  = (const float*)d_in[20];
    const float* bao = (const float*)d_in[21];
    const float* nq  = (const float*)d_in[22];
    const float* nk  = (const float*)d_in[23];
    const float* naq = (const float*)d_in[24];
    const float* nak = (const float*)d_in[25];

    const size_t perT = (size_t)HEADS * SEQ * HD;   // 7,864,320 bf16 elements
    __bf16* Qb = (__bf16*)d_ws;
    __bf16* Kb = Qb + perT;
    __bf16* Vb = Kb + perT;
    __bf16* Ab = Vb + perT;                          // attn output, [seq][3072] bf16

    float* outImg = (float*)d_out;
    float* outTxt = outImg + (size_t)S_IMG * DIM;

    dim3 blk(256);
    dim3 gImg(S_IMG / TM, HEADS), gTxt(S_TXT / TM, HEADS);

    // img-stream projections (concat positions start at S_TXT)
    gemm_proj<<<gImg, blk, 0, stream>>>(hidden, nullptr, Wq, bq, 1, Qb, nullptr, img_cos, img_sin, nq, S_TXT);
    gemm_proj<<<gImg, blk, 0, stream>>>(hidden, nullptr, Wk, bk, 1, Kb, nullptr, img_cos, img_sin, nk, S_TXT);
    gemm_proj<<<gImg, blk, 0, stream>>>(hidden, nullptr, Wv, bv, 2, Vb, nullptr, nullptr, nullptr, nullptr, S_TXT);
    // txt-stream projections (concat positions start at 0)
    gemm_proj<<<gTxt, blk, 0, stream>>>(enc, nullptr, Waq, baq, 1, Qb, nullptr, txt_cos, txt_sin, naq, 0);
    gemm_proj<<<gTxt, blk, 0, stream>>>(enc, nullptr, Wak, bak, 1, Kb, nullptr, txt_cos, txt_sin, nak, 0);
    gemm_proj<<<gTxt, blk, 0, stream>>>(enc, nullptr, Wav, bav, 2, Vb, nullptr, nullptr, nullptr, nullptr, 0);
    // flash attention over the concatenated sequence
    attn_fa<<<dim3(SEQ / BR, HEADS), blk, 0, stream>>>(Qb, Kb, Vb, Ab);
    // output projections: img rows [512,2560) -> outImg; txt rows [0,512) -> outTxt
    gemm_proj<<<gImg, blk, 0, stream>>>(nullptr, Ab + (size_t)S_TXT * DIM, Wo, bo, 3, nullptr, outImg, nullptr, nullptr, nullptr, 0);
    gemm_proj<<<gTxt, blk, 0, stream>>>(nullptr, Ab, Wao, bao, 3, nullptr, outTxt, nullptr, nullptr, nullptr, 0);
}
